// PrototypeBank_55688545960231
// MI455X (gfx1250) — matrix operand order repaired and checked by reading, not compiled, not run on hardware
//
#include <hip/hip_runtime.h>

#define B_SAMPLES 65536
#define FDIM      512
#define NCLS      64
#define MPROT     4
#define KM        256          // NCLS*MPROT
#define INV_TEMP  0.1f         // 1/10.0
#define INV_SEPS  20.0f        // 1/0.05
#define APAD      520          // LDS row stride in halves (1040B = 65*16B, bank-conflict free)

typedef _Float16 v8h  __attribute__((ext_vector_type(8)));
typedef _Float16 v16h __attribute__((ext_vector_type(16)));
typedef float    v8f  __attribute__((ext_vector_type(8)));
typedef float    v4f  __attribute__((ext_vector_type(4)));

__device__ __forceinline__ float wave_sum32(float v) {
#pragma unroll
  for (int o = 16; o > 0; o >>= 1) v += __shfl_xor(v, o, 32);
  return v;
}
__device__ __forceinline__ unsigned wave_or32(unsigned v) {
#pragma unroll
  for (int o = 16; o > 0; o >>= 1) v |= __shfl_xor(v, o, 32);
  return v;
}

// ---------------- init: zero accumulators, gamma = 1 ----------------
__global__ void k_init(float* __restrict__ counts, float* __restrict__ col,
                       float* __restrict__ gamma, float* __restrict__ pcls) {
  int t = threadIdx.x;                 // 256 threads
  if (t < NCLS) { counts[t] = 0.f; pcls[t] = 0.f; }
  col[t]   = 0.f;
  gamma[t] = 1.f;
}

// ---------------- normalize prototypes -> f16 ----------------
__global__ void __launch_bounds__(32) k_proto(const float* __restrict__ proto,
                                              _Float16* __restrict__ pn) {
  const int row  = blockIdx.x;         // 0..255
  const int lane = threadIdx.x;        // wave32
  const float4* pr = (const float4*)(proto + (size_t)row * FDIM) + lane * 4;
  float t16[16];
  float ss = 0.f;
#pragma unroll
  for (int j = 0; j < 4; ++j) {
    float4 v = pr[j];
    t16[4*j+0] = v.x; t16[4*j+1] = v.y; t16[4*j+2] = v.z; t16[4*j+3] = v.w;
    ss += v.x*v.x + v.y*v.y + v.z*v.z + v.w*v.w;
  }
  ss = wave_sum32(ss);
  const float s = 1.f / fmaxf(sqrtf(ss), 1e-12f);
  v8h h0, h1;
#pragma unroll
  for (int i = 0; i < 8; ++i) { h0[i] = (_Float16)(t16[i] * s); h1[i] = (_Float16)(t16[8+i] * s); }
  _Float16* dst = pn + (size_t)row * FDIM + lane * 16;
  *(v8h*)dst = h0; *(v8h*)(dst + 8) = h1;
}

// ---------------- class counts ----------------
__global__ void k_counts(const int* __restrict__ labels, float* __restrict__ counts) {
  __shared__ float h[NCLS];
  const int t = threadIdx.x;
  if (t < NCLS) h[t] = 0.f;
  __syncthreads();
  const int i = blockIdx.x * 256 + t;
  atomicAdd(&h[labels[i]], 1.0f);
  __syncthreads();
  if (t < NCLS) atomicAdd(&counts[t], h[t]);
}

// ---------------- fused normalize + WMMA cosine-sim + own-class gather ----------------
__global__ void __launch_bounds__(32) k_sim(const float* __restrict__ z,
                                            const _Float16* __restrict__ pn,
                                            const int* __restrict__ labels,
                                            float* __restrict__ simlab) {
  __shared__ _Float16 sA[16 * APAD];   // 16 rows of normalized z in f16, padded
  const int lane = threadIdx.x;
  const int base = blockIdx.x * 16;

  // labels for the 16 rows of this tile; mask of which 16-col N-tiles are needed
  const int lab = (lane < 16) ? labels[base + lane] : 0;
  unsigned nm = (lane < 16) ? (1u << (lab >> 2)) : 0u;   // class c lives in N-tile c/4
  nm = wave_or32(nm);

  // stage: stream z rows (non-temporal, read-once), L2-normalize in f32, store f16 to LDS
  for (int r = 0; r < 16; ++r) {
    const v4f* zr = (const v4f*)(z + (size_t)(base + r) * FDIM) + lane * 4;
    float t16[16];
    float ss = 0.f;
#pragma unroll
    for (int j = 0; j < 4; ++j) {
      v4f v = __builtin_nontemporal_load(zr + j);
      t16[4*j+0] = v[0]; t16[4*j+1] = v[1]; t16[4*j+2] = v[2]; t16[4*j+3] = v[3];
      ss += v[0]*v[0] + v[1]*v[1] + v[2]*v[2] + v[3]*v[3];
    }
    ss = wave_sum32(ss);
    const float s = 1.f / fmaxf(sqrtf(ss), 1e-12f);
    v8h h0, h1;
#pragma unroll
    for (int i = 0; i < 8; ++i) { h0[i] = (_Float16)(t16[i] * s); h1[i] = (_Float16)(t16[8+i] * s); }
    _Float16* dst = &sA[r * APAD + lane * 16];
    *(v8h*)dst = h0; *(v8h*)(dst + 8) = h1;
  }
  __syncthreads();

  const int m = lane & 15;             // A-row / B-col index for this lane
  const int h = lane >> 4;             // lane half
  // A fragment (16x32 f16): lane half h holds contiguous K = {kb+16h .. kb+16h+15} in v[0..15]
  const _Float16* aRow = &sA[m * APAD + 16 * h];
  // B fragment (32x16 f16): lane half h holds contiguous K = {kb+16h .. kb+16h+15} of column m
  //   (same contiguous K order as A, not the ISA table order; valid because A and B match)

  for (int nt = 0; nt < 16; ++nt) {
    if (!((nm >> nt) & 1u)) continue;  // wave-uniform skip: EXEC stays full for WMMA
    const _Float16* bRow = pn + (size_t)(nt * 16 + m) * FDIM + 16 * h;
    v8f acc = {};
#pragma unroll
    for (int ks = 0; ks < 16; ++ks) {
      const int kb = ks * 32;
      v8h a0 = *(const v8h*)(aRow + kb);        // K = kb + 16h .. +8
      v8h a1 = *(const v8h*)(aRow + kb + 8);    // K = kb + 16h + 8 .. +8
      v16h A  = __builtin_shufflevector(a0, a1, 0,1,2,3,4,5,6,7,8,9,10,11,12,13,14,15);
      v16h Bf = *(const v16h*)(bRow + kb);      // K = kb + 16h .. +16 (contiguous, 32B aligned)
      acc = __builtin_amdgcn_wmma_f32_16x16x32_f16(false, A, false, Bf,
                                                   (short)0, acc, false, false);
    }
    // scatter own-class block: acc[v] = sim[M = v + 8h][N = m], global col = 16*nt + m
#pragma unroll
    for (int v = 0; v < 8; ++v) {
      const int r  = v + 8 * h;
      const int lr = __shfl(lab, r, 32);
      if ((lr >> 2) == nt) {
        const int c0 = (lr & 3) * 4;
        const int n  = m - c0;
        if ((unsigned)n < 4u)
          simlab[(size_t)(base + r) * 4 + n] = acc[v];
      }
    }
  }
}

// ---------------- Sinkhorn round: col[k,m] = seg-sum of rownorm(e * gamma) ----------------
__global__ void k_colsum(const float* __restrict__ simlab, const int* __restrict__ labels,
                         const float* __restrict__ gamma, float* __restrict__ col) {
  __shared__ float lc[KM];
  const int t = threadIdx.x;           // 256
  lc[t] = 0.f;
  __syncthreads();
  const int i = blockIdx.x * 256 + t;
  const float4 s = ((const float4*)simlab)[i];
  const int lab = labels[i];
  const float4 g = ((const float4*)gamma)[lab];
  const float q0 = __expf(s.x * INV_SEPS) * g.x;
  const float q1 = __expf(s.y * INV_SEPS) * g.y;
  const float q2 = __expf(s.z * INV_SEPS) * g.z;
  const float q3 = __expf(s.w * INV_SEPS) * g.w;
  const float inv = 1.f / (q0 + q1 + q2 + q3 + 1e-12f);
  atomicAdd(&lc[lab * 4 + 0], q0 * inv);
  atomicAdd(&lc[lab * 4 + 1], q1 * inv);
  atomicAdd(&lc[lab * 4 + 2], q2 * inv);
  atomicAdd(&lc[lab * 4 + 3], q3 * inv);
  __syncthreads();
  atomicAdd(&col[t], lc[t]);
}

// ---------------- Sinkhorn gamma update + reset col ----------------
__global__ void k_gamma(const float* __restrict__ counts, float* __restrict__ col,
                        float* __restrict__ gamma) {
  const int t = threadIdx.x;           // 256
  const int k = t >> 2;
  const float beta = (counts[k] * (1.0f / MPROT)) / (col[t] + 1e-12f);
  gamma[t] *= beta;
  col[t] = 0.f;
}

// ---------------- per-class loss accumulation ----------------
__global__ void k_loss(const float* __restrict__ simlab, const int* __restrict__ labels,
                       const float* __restrict__ gamma, float* __restrict__ pcls) {
  __shared__ float lp[NCLS];
  const int t = threadIdx.x;           // 256
  if (t < NCLS) lp[t] = 0.f;
  __syncthreads();
  const int i = blockIdx.x * 256 + t;
  const float4 s = ((const float4*)simlab)[i];
  const int lab = labels[i];
  const float4 g = ((const float4*)gamma)[lab];
  float q0 = __expf(s.x * INV_SEPS) * g.x;
  float q1 = __expf(s.y * INV_SEPS) * g.y;
  float q2 = __expf(s.z * INV_SEPS) * g.z;
  float q3 = __expf(s.w * INV_SEPS) * g.w;
  const float inv = 1.f / (q0 + q1 + q2 + q3 + 1e-12f);
  q0 *= inv; q1 *= inv; q2 *= inv; q3 *= inv;
  const float l0 = s.x * INV_TEMP, l1 = s.y * INV_TEMP, l2 = s.z * INV_TEMP, l3 = s.w * INV_TEMP;
  const float mx = fmaxf(fmaxf(l0, l1), fmaxf(l2, l3));
  const float se = __expf(l0 - mx) + __expf(l1 - mx) + __expf(l2 - mx) + __expf(l3 - mx);
  const float lse = mx + __logf(se);
  const float ps = q0 * (lse - l0) + q1 * (lse - l1) + q2 * (lse - l2) + q3 * (lse - l3);
  atomicAdd(&lp[lab], ps);
  __syncthreads();
  if (t < NCLS) atomicAdd(&pcls[t], lp[t]);
}

// ---------------- final scalar ----------------
__global__ void k_final(const float* __restrict__ counts, const float* __restrict__ pcls,
                        float* __restrict__ out) {
  __shared__ float sv[NCLS];
  __shared__ float sn[NCLS];
  const int t = threadIdx.x;           // 64
  const float c = counts[t];
  sv[t] = (c > 0.f) ? pcls[t] / fmaxf(c, 1.f) : 0.f;
  sn[t] = (c > 0.f) ? 1.f : 0.f;
  __syncthreads();
  if (t == 0) {
    float sum = 0.f, n = 0.f;
    for (int j = 0; j < NCLS; ++j) { sum += sv[j]; n += sn[j]; }
    out[0] = sum / fmaxf(n, 1.f);
  }
}

extern "C" void kernel_launch(void* const* d_in, const int* in_sizes, int n_in,
                              void* d_out, int out_size, void* d_ws, size_t ws_size,
                              hipStream_t stream) {
  const float* z      = (const float*)d_in[0];   // (B, 512) f32
  const float* proto  = (const float*)d_in[1];   // (256, 512) f32
  const int*   labels = (const int*)d_in[2];     // (B,) int
  float*       out    = (float*)d_out;           // scalar

  char* w = (char*)d_ws;
  _Float16* pn     = (_Float16*)(w);                         // 256*512*2   = 262144 B
  float*    simlab = (float*)(w + 262144);                   // B*4*4       = 1048576 B
  float*    counts = (float*)(w + 262144 + 1048576);         // 64 f
  float*    col    = (float*)(w + 262144 + 1048576 + 256);   // 256 f
  float*    gamma  = (float*)(w + 262144 + 1048576 + 1280);  // 256 f
  float*    pcls   = (float*)(w + 262144 + 1048576 + 2304);  // 64 f

  k_init  <<<1, 256, 0, stream>>>(counts, col, gamma, pcls);
  k_proto <<<KM, 32, 0, stream>>>(proto, pn);
  k_counts<<<B_SAMPLES / 256, 256, 0, stream>>>(labels, counts);
  k_sim   <<<B_SAMPLES / 16, 32, 0, stream>>>(z, pn, labels, simlab);
  for (int it = 0; it < 3; ++it) {
    k_colsum<<<B_SAMPLES / 256, 256, 0, stream>>>(simlab, labels, gamma, col);
    k_gamma <<<1, 256, 0, stream>>>(counts, col, gamma);
  }
  k_loss  <<<B_SAMPLES / 256, 256, 0, stream>>>(simlab, labels, gamma, pcls);
  k_final <<<1, NCLS, 0, stream>>>(counts, pcls, out);
}
